// FastCMIF_4123168604532
// MI455X (gfx1250) — 8-bit matrix operand order repaired and checked by reading, compiled, not run on hardware
//
#include <hip/hip_runtime.h>
#include <math.h>


typedef __attribute__((ext_vector_type(8))) int v8i;

#define NBINS    16
#define IMH      384
#define IMW      384
#define TPH      48
#define TPW      48
#define PAD      24
#define KTOT     (TPH * TPW)
#define KCHUNKS  (KTOT / 64)
#define NPIX     (IMH * IMW)
#define PROWS    432
#define PDW      16
#define ROWDW    (NBINS * PDW)

#define WS_MM_IM   0
#define WS_MM_T    16
#define WS_TBINS   32
#define WS_APACK   4640
#define WS_PLANES  (4640 + 73728)

__global__ __launch_bounds__(256)
void cmif_minmax_kernel(const float* __restrict__ x, int n_per_batch,
                        float* __restrict__ mm) {
    const int b = blockIdx.x;
    const float* p = x + (size_t)b * n_per_batch;
    float mn = INFINITY, mx = -INFINITY;
    for (int i = threadIdx.x; i < n_per_batch; i += 256) {
        float v = p[i];
        mn = fminf(mn, v);
        mx = fmaxf(mx, v);
    }
    __shared__ float smn[256], smx[256];
    smn[threadIdx.x] = mn;
    smx[threadIdx.x] = mx;
    __syncthreads();
    for (int s = 128; s > 0; s >>= 1) {
        if (threadIdx.x < s) {
            smn[threadIdx.x] = fminf(smn[threadIdx.x], smn[threadIdx.x + s]);
            smx[threadIdx.x] = fmaxf(smx[threadIdx.x], smx[threadIdx.x + s]);
        }
        __syncthreads();
    }
    if (threadIdx.x == 0) {
        mm[b * 2 + 0] = smn[0];
        mm[b * 2 + 1] = smx[0];
    }
}

__device__ __forceinline__ int cmif_bin(float v, float mn, float mx) {
    int q = (int)floorf((v - mn) / (mx - mn) * (float)(NBINS - 1));
    return q < 0 ? 0 : (q > NBINS - 1 ? NBINS - 1 : q);
}

__global__ __launch_bounds__(256)
void cmif_quantize_t_kernel(const float* __restrict__ x,
                            const float* __restrict__ mm,
                            unsigned char* __restrict__ q) {
    int idx = blockIdx.x * 256 + threadIdx.x;
    if (idx >= 2 * KTOT) return;
    int b = idx / KTOT;
    q[idx] = (unsigned char)cmif_bin(x[idx], mm[b * 2 + 0], mm[b * 2 + 1]);
}

__global__ __launch_bounds__(256)
void cmif_packA_kernel(const unsigned char* __restrict__ tbins,
                       unsigned int* __restrict__ apack) {
    int idx = blockIdx.x * 256 + threadIdx.x;
    if (idx >= 2 * KCHUNKS * 32) return;
    int lane = idx & 31;
    int ch   = (idx >> 5) % KCHUNKS;
    int b    = idx / (KCHUNKS * 32);
    int m    = lane & 15;
    int half = lane >> 4;
    const unsigned char* tB = tbins + (size_t)b * KTOT;
#pragma unroll
    for (int p = 0; p < 8; ++p) {
        int ka = ch * 64 + ((p >> 2) << 5) + (half << 4) + ((p & 3) << 2);
        unsigned int w = 0;
#pragma unroll
        for (int e = 0; e < 4; ++e)
            w |= (unsigned int)(tB[ka + e] == (unsigned char)m) << (8 * e);
        apack[(size_t)idx * 8 + p] = w;
    }
}

__global__ __launch_bounds__(256)
void cmif_bitplane_kernel(const float* __restrict__ im,
                          const float* __restrict__ mm,
                          unsigned int* __restrict__ planes) {
    int idx = blockIdx.x * 256 + threadIdx.x;
    int d   = idx & 15;
    int bin = (idx >> 4) & 15;
    int rem = idx >> 8;
    int pr  = rem % PROWS;
    int b   = rem / PROWS;
    int gy  = pr - PAD;
    unsigned int w = 0;
    if (gy >= 0 && gy < IMH) {
        float mn = mm[b * 2 + 0], mx = mm[b * 2 + 1];
        const float* row = im + (size_t)b * NPIX + (size_t)gy * IMW;
        for (int t = 0; t < 32; ++t) {
            int gx = d * 32 + t - PAD;
            if (gx >= 0 && gx < IMW)
                w |= (unsigned int)(cmif_bin(row[gx], mn, mx) == bin) << t;
        }
    }
    planes[idx] = w;
}

__global__ __launch_bounds__(256)
void cmif_mi_kernel(const unsigned int* __restrict__ apack,
                    const unsigned int* __restrict__ planes,
                    float* __restrict__ out) {
    const int tid  = threadIdx.x;
    const int wave = tid >> 5;
    const int lane = tid & 31;

    const int pix = blockIdx.x * 8 + wave;
    const int b   = pix / NPIX;
    const int rem = pix - b * NPIX;
    const int y   = rem / IMW;
    const int x   = rem - y * IMW;

    const int bin  = lane & 15;
    const int half = lane >> 4;

    const unsigned int* ap = apack + ((size_t)b * KCHUNKS * 32 + lane) * 8;
    const unsigned int* pl = planes + (size_t)b * PROWS * ROWDW;

    const unsigned int* Pe[3];
    int she[3];
    {
        const int v0[3] = {0, 2, 1};
        const int v1[3] = {1, 0, 2};
#pragma unroll
        for (int e = 0; e < 3; ++e) {
            int vi  = half ? v1[e] : v0[e];
            int duE = (e == 0) ? 0 : ((e == 1) ? half : 1);
            int bx  = x + (vi << 4);
            Pe[e]  = pl + ((size_t)(y + duE) * ROWDW + (bin << 4) + (bx >> 5));
            she[e] = bx & 31;
        }
    }

    v8i acc = {};
    v8i bm;
#pragma unroll
    for (int t = 0; t < 72; ++t) {
        const int s = t / 3;
        const int e = t - 3 * s;
        const int r = t & 1;
        const unsigned int* P = Pe[e] + (size_t)s * 2 * ROWDW;
        unsigned long long w =
            ((unsigned long long)P[1] << 32) | (unsigned long long)P[0];
        unsigned int lo = (unsigned int)(w >> she[e]);
#pragma unroll
        for (int g = 0; g < 4; ++g)
            bm[r * 4 + g] =
                (int)((((lo >> (4 * g)) & 0xFu) * 0x00204081u) & 0x01010101u);
        if (r == 1) {
            v8i a = *(const v8i*)(ap + (size_t)(t >> 1) * 32 * 8);
            acc = __builtin_amdgcn_wmma_i32_16x16x64_iu8(false, a, false, bm,
                                                         acc, false, false);
        }
    }

    int rs[8];
#pragma unroll
    for (int r = 0; r < 8; ++r) {
        int s = acc[r];
        s += __shfl_xor(s, 1);
        s += __shfl_xor(s, 2);
        s += __shfl_xor(s, 4);
        s += __shfl_xor(s, 8);
        rs[r] = s;
    }
    int cs = 0;
#pragma unroll
    for (int r = 0; r < 8; ++r) cs += acc[r];
    cs += __shfl_xor(cs, 16);

    const int rcnt = (y + PAD < IMH ? y + PAD : IMH) - (y - PAD > 0 ? y - PAD : 0);
    const int ccnt = (x + PAD < IMW ? x + PAD : IMW) - (x - PAD > 0 ? x - PAD : 0);
    const int n = rcnt * ccnt;

    const float logN = __logf((float)n);
    const float logC = (cs > 0) ? __logf((float)cs) : 0.0f;
    float mi = 0.0f;
#pragma unroll
    for (int r = 0; r < 8; ++r) {
        int c = acc[r];
        float lc = (c > 0) ? __logf((float)c) : 0.0f;
        float lr = (rs[r] > 0) ? __logf((float)rs[r]) : 0.0f;
        mi += (c > 0) ? (float)c * (lc + logN - lr - logC) : 0.0f;
    }
    mi += __shfl_xor(mi, 1);
    mi += __shfl_xor(mi, 2);
    mi += __shfl_xor(mi, 4);
    mi += __shfl_xor(mi, 8);
    mi += __shfl_xor(mi, 16);

    if (lane == 0) out[pix] = mi / (float)n;
}

extern "C" void kernel_launch(void* const* d_in, const int* in_sizes, int n_in,
                              void* d_out, int out_size, void* d_ws, size_t ws_size,
                              hipStream_t stream) {
    const float* im = (const float*)d_in[0];
    const float* tm = (const float*)d_in[1];
    float* out = (float*)d_out;

    char* ws = (char*)d_ws;
    float*         mmIm   = (float*)(ws + WS_MM_IM);
    float*         mmT    = (float*)(ws + WS_MM_T);
    unsigned char* tbins  = (unsigned char*)(ws + WS_TBINS);
    unsigned int*  apack  = (unsigned int*)(ws + WS_APACK);
    unsigned int*  planes = (unsigned int*)(ws + WS_PLANES);

    cmif_minmax_kernel<<<2, 256, 0, stream>>>(im, NPIX, mmIm);
    cmif_minmax_kernel<<<2, 256, 0, stream>>>(tm, KTOT, mmT);

    cmif_quantize_t_kernel<<<(2 * KTOT + 255) / 256, 256, 0, stream>>>(
        tm, mmT, tbins);

    cmif_packA_kernel<<<(2 * KCHUNKS * 32 + 255) / 256, 256, 0, stream>>>(
        tbins, apack);

    cmif_bitplane_kernel<<<(2 * PROWS * NBINS * PDW) / 256, 256, 0, stream>>>(
        im, mmIm, planes);

    cmif_mi_kernel<<<(2 * NPIX) / 8, 256, 0, stream>>>(apack, planes, out);
}
